// GraphCheb_71064528880189
// MI455X (gfx1250) — matrix operand order repaired and checked by reading, not compiled, not run on hardware
//
#include <hip/hip_runtime.h>
#include <hip/hip_bf16.h>
#include <cstddef>

#define N_NODES   131072
#define N_EDGES   2097152
#define N_GRAPHS  64
#define F_IN      64
#define HIDDEN    128
#define K_CHEB    5
#define NODES_PER_GRAPH (N_NODES / N_GRAPHS)

typedef __attribute__((ext_vector_type(16))) __bf16 v16bf;
typedef __attribute__((ext_vector_type(8)))  float v8f;

// ---------- f32 -> packed bf16x2 conversion ----------
#if __has_builtin(__builtin_amdgcn_cvt_pk_bf16_f32)
typedef __attribute__((ext_vector_type(2))) __bf16 v2bf;
__device__ __forceinline__ unsigned int pack2bf(float a, float b) {
    v2bf r = __builtin_amdgcn_cvt_pk_bf16_f32(a, b);
    union { v2bf v; unsigned int u; } x; x.v = r;
    return x.u;
}
#else
__device__ __forceinline__ unsigned int f2bf_u(float f) {
    union { float f; unsigned int u; } x; x.f = f;
    return (x.u + 0x7FFFu + ((x.u >> 16) & 1u)) >> 16;       // RNE
}
__device__ __forceinline__ unsigned int pack2bf(float a, float b) {
    return f2bf_u(a) | (f2bf_u(b) << 16);
}
#endif
__device__ __forceinline__ unsigned short f2bf16(float a) {
    return (unsigned short)(pack2bf(a, 0.0f) & 0xFFFFu);
}

// ---------- tiny elementwise kernels ----------
__global__ void zero_f32(float* __restrict__ p, size_t n) {
    size_t i = (size_t)blockIdx.x * blockDim.x + threadIdx.x;
    size_t stride = (size_t)gridDim.x * blockDim.x;
    for (; i < n; i += stride) p[i] = 0.0f;
}

__global__ void fill_f32(float* __restrict__ p, float v, size_t n) {
    size_t i = (size_t)blockIdx.x * blockDim.x + threadIdx.x;
    size_t stride = (size_t)gridDim.x * blockDim.x;
    for (; i < n; i += stride) p[i] = v;
}

__global__ void neg_copy_f32(const float* __restrict__ src, float* __restrict__ dst, size_t n) {
    size_t i = (size_t)blockIdx.x * blockDim.x + threadIdx.x;
    size_t stride = (size_t)gridDim.x * blockDim.x;
    for (; i < n; i += stride) dst[i] = -src[i];
}

// ---------- one-time weight conversion: W[k][kd][n] f32 -> Wbf[k][n][kd] bf16 ----------
__global__ void convert_wT(const float* __restrict__ W, unsigned short* __restrict__ Wbf,
                           int Cin) {
    int idx = blockIdx.x * blockDim.x + threadIdx.x;
    int total = K_CHEB * Cin * HIDDEN;
    if (idx >= total) return;
    int k  = idx / (Cin * HIDDEN);
    int r  = idx % (Cin * HIDDEN);
    int n  = r / Cin;            // output column 0..127
    int kd = r % Cin;            // input dim
    float v = W[(size_t)k * Cin * HIDDEN + (size_t)kd * HIDDEN + n];
    Wbf[idx] = f2bf16(v);
}

// ---------- degree / norm ----------
__global__ void count_deg(const int* __restrict__ row, float* __restrict__ deg, int E) {
    int e = blockIdx.x * blockDim.x + threadIdx.x;
    if (e < E) unsafeAtomicAdd(&deg[row[e]], 1.0f);
}

__global__ void finalize_dis(float* __restrict__ d, int n) {
    int i = blockIdx.x * blockDim.x + threadIdx.x;
    if (i < n) {
        float deg = d[i];
        d[i] = (deg > 0.0f) ? __frsqrt_rn(fmaxf(deg, 1.0f)) : 0.0f;
    }
}

__global__ void make_norm(const int* __restrict__ row, const int* __restrict__ col,
                          const float* __restrict__ dis, float* __restrict__ norm, int E) {
    int e = blockIdx.x * blockDim.x + threadIdx.x;
    if (e < E) norm[e] = -dis[row[e]] * dis[col[e]];
}

// ---------- sparse propagation: Tout[col[e]] += scale*norm[e]*Tin[row[e]] ----------
__global__ __launch_bounds__(256)
void prop_scatter(const int* __restrict__ row, const int* __restrict__ col,
                  const float* __restrict__ norm,
                  const float* __restrict__ Tin, float* __restrict__ Tout,
                  int E, int C, float scale) {
    int e    = blockIdx.x * 8 + (threadIdx.x >> 5);
    int lane = threadIdx.x & 31;
    if (e >= E) return;
    int   r = row[e];
    int   c = col[e];
    float w = norm[e] * scale;
    int f = lane * 4;
    if (f < C) {
        const float4 v = *(const float4*)(Tin + (size_t)r * C + f);
        float* o = Tout + (size_t)c * C + f;
        unsafeAtomicAdd(o + 0, w * v.x);
        unsafeAtomicAdd(o + 1, w * v.y);
        unsafeAtomicAdd(o + 2, w * v.z);
        unsafeAtomicAdd(o + 3, w * v.w);
    }
}

// ---------- WMMA GEMM: acc[M x 128] (+)= T[M x KDIM] @ Wbf[KDIM x 128] ----------
// Wbf is pre-converted bf16 in transposed [n][kdim] layout.
#define GEMM_FLAG_BIAS 1
#define GEMM_FLAG_RELU 2
#define TM 64
#define TN 128

union ABfrag { v16bf v; unsigned int u[8]; };

template <int KDIM, int FLAGS>
__global__ __launch_bounds__(256)
void gemm_cheb(const float* __restrict__ T, const unsigned short* __restrict__ Wbf,
               const float* __restrict__ bias, float* __restrict__ acc) {
    // bf16 tiles, packed uint pairs; pitch = 16 uints (32 bf16) per row
    __shared__ __align__(16) unsigned int lA[TM * 16];   // [m][k]
    __shared__ __align__(16) unsigned int lB[TN * 16];   // [n][k]

    const int tid   = threadIdx.x;
    const int lane  = tid & 31;
    const int wave  = tid >> 5;
    const int wm    = wave & 3;          // 4 M sub-tiles of 16
    const int wn    = wave >> 2;         // 2 N sub-tiles of 64
    const int mOff  = wm * 16;
    const int nOff  = wn * 64;
    const int mBase = blockIdx.x * TM;

    const int nLane  = lane & 15;        // N within subtile; also A-row lane
    const int rowSel = (lane >> 4) * 8;  // C/D: VGPR v -> M = v + rowSel

    v8f c[4];
    if (FLAGS & GEMM_FLAG_BIAS) {
#pragma unroll
        for (int nt = 0; nt < 4; ++nt) {
            float bv = bias[nOff + nt * 16 + nLane];
            c[nt] = (v8f){bv, bv, bv, bv, bv, bv, bv, bv};
        }
    } else {
#pragma unroll
        for (int nt = 0; nt < 4; ++nt) {
            int n = nOff + nt * 16 + nLane;
#pragma unroll
            for (int v = 0; v < 8; ++v) {
                size_t r = (size_t)(mBase + mOff + rowSel + v);
                c[nt][v] = acc[r * HIDDEN + n];
            }
        }
    }

    // staging coordinates (8 contiguous elements per thread -> one b128 DS store)
    const int aM  = tid >> 2;            // 0..63
    const int aKq = tid & 3;             // k-octet 0..3

#pragma unroll
    for (int kk = 0; kk < KDIM; kk += 32) {
        __syncthreads();
        // ---- stage A tile: 64 x 32 f32 -> bf16, [m][k] ----
        {
            const float* src = T + (size_t)(mBase + aM) * KDIM + kk + aKq * 8;
            float4 f0 = *(const float4*)(src);
            float4 f1 = *(const float4*)(src + 4);
            uint4 p;
            p.x = pack2bf(f0.x, f0.y);
            p.y = pack2bf(f0.z, f0.w);
            p.z = pack2bf(f1.x, f1.y);
            p.w = pack2bf(f1.z, f1.w);
            ((uint4*)lA)[aM * 4 + aKq] = p;
        }
        // ---- stage B tile: pure b128 copy (weights pre-converted, pre-transposed) ----
#pragma unroll
        for (int it = 0; it < 2; ++it) {
            int idx = it * 256 + tid;            // 0..511
            int bKq = idx >> 7;                  // k-octet 0..3
            int bN  = idx & 127;                 // n
            uint4 p = *(const uint4*)(Wbf + (size_t)bN * KDIM + kk + bKq * 8);
            ((uint4*)lB)[bN * 4 + bKq] = p;
        }
        __syncthreads();

        // A fragment (16x32 bf16): lanes 0-15 -> K 0..15, lanes 16-31 -> K 16..31
        ABfrag a;
        {
            int m    = mOff + nLane;
            int hsel = (lane >> 4) * 8;
#pragma unroll
            for (int v = 0; v < 8; ++v) {
                a.u[v] = lA[m * 16 + hsel + v];
            }
        }
        // Preload all four B fragments so DS loads issue as a clause.
        ABfrag b[4];
        const int kb = (lane >> 4) * 16;
#pragma unroll
        for (int nt = 0; nt < 4; ++nt) {
            int n = nOff + nt * 16 + nLane;
#pragma unroll
            for (int v = 0; v < 8; ++v) b[nt].u[v] = lB[n * 16 + (kb >> 1) + v];
        }
#pragma unroll
        for (int nt = 0; nt < 4; ++nt) {
            c[nt] = __builtin_amdgcn_wmma_f32_16x16x32_bf16(
                        false, a.v, false, b[nt].v, (short)0, c[nt], false, false);
        }
    }

#pragma unroll
    for (int nt = 0; nt < 4; ++nt) {
        int n = nOff + nt * 16 + nLane;
#pragma unroll
        for (int v = 0; v < 8; ++v) {
            size_t r = (size_t)(mBase + mOff + rowSel + v);
            float val = c[nt][v];
            if (FLAGS & GEMM_FLAG_RELU) val = fmaxf(val, 0.0f);
            acc[r * HIDDEN + n] = val;
        }
    }
}

// ---------- global max pool, 2-level: block-partial max then float atomic-max ----------
__global__ void pool_max_partial(const float* __restrict__ h, float* __restrict__ gmax) {
    int g     = blockIdx.x;            // graph
    int chunk = blockIdx.y;            // 16 chunks of 128 nodes
    int f     = threadIdx.x;           // 128 features
    const float* base = h + ((size_t)g * NODES_PER_GRAPH + (size_t)chunk * 128) * HIDDEN + f;
    float m = -INFINITY;
    for (int i = 0; i < 128; ++i) m = fmaxf(m, base[(size_t)i * HIDDEN]);
    unsafeAtomicMax(&gmax[g * HIDDEN + f], m);
}

// ---------- final linear head: [64 x 128] @ [128 x 2] + b ----------
__global__ void head_linear(const float* __restrict__ gmax, const float* __restrict__ Wlin,
                            const float* __restrict__ blin, float* __restrict__ out) {
    int t = threadIdx.x;               // 128 threads = 64 graphs * 2 classes
    int g = t >> 1, cls = t & 1;
    float s = blin[cls];
    for (int f = 0; f < HIDDEN; ++f) s += gmax[g * HIDDEN + f] * Wlin[f * 2 + cls];
    out[g * 2 + cls] = s;
}

// ---------- host orchestration ----------
extern "C" void kernel_launch(void* const* d_in, const int* in_sizes, int n_in,
                              void* d_out, int out_size, void* d_ws, size_t ws_size,
                              hipStream_t stream) {
    const float* x    = (const float*)d_in[0];
    const int*   row  = (const int*)d_in[1];
    const int*   col  = row + N_EDGES;
    const float* W0   = (const float*)d_in[3];
    const float* b0   = (const float*)d_in[4];
    const float* W1   = (const float*)d_in[5];
    const float* b1   = (const float*)d_in[6];
    const float* W2   = (const float*)d_in[7];
    const float* b2   = (const float*)d_in[8];
    const float* Wlin = (const float*)d_in[9];
    const float* blin = (const float*)d_in[10];
    float* out = (float*)d_out;

    // workspace carving (256B aligned)
    char* w = (char*)d_ws;
    size_t off = 0;
    auto carve = [&](size_t bytes) -> char* {
        char* p = w + off;
        off = (off + bytes + 255) & ~(size_t)255;
        return p;
    };
    float* dis   = (float*)carve((size_t)N_NODES * 4);
    float* norm  = (float*)carve((size_t)N_EDGES * 4);
    float* gmax  = (float*)carve((size_t)N_GRAPHS * HIDDEN * 4);
    unsigned short* Wbf0 = (unsigned short*)carve((size_t)K_CHEB * F_IN   * HIDDEN * 2);
    unsigned short* Wbf1 = (unsigned short*)carve((size_t)K_CHEB * HIDDEN * HIDDEN * 2);
    unsigned short* Wbf2 = (unsigned short*)carve((size_t)K_CHEB * HIDDEN * HIDDEN * 2);
    const size_t nodeBuf = (size_t)N_NODES * HIDDEN * 4;
    float* bufA = (float*)carve(nodeBuf);   // T rotation
    float* bufB = (float*)carve(nodeBuf);
    float* bufC = (float*)carve(nodeBuf);
    float* bufD = (float*)carve(nodeBuf);   // acc / h ping
    float* bufE = (float*)carve(nodeBuf);   // acc / h pong

    const int EW_BLOCKS = 4096;
    const int gemmGrid  = N_NODES / TM;
    const int propGrid  = N_EDGES / 8;

    // ----- one-time weight conversion to transposed bf16 -----
    convert_wT<<<(K_CHEB * F_IN   * HIDDEN + 255) / 256, 256, 0, stream>>>(W0, Wbf0, F_IN);
    convert_wT<<<(K_CHEB * HIDDEN * HIDDEN + 255) / 256, 256, 0, stream>>>(W1, Wbf1, HIDDEN);
    convert_wT<<<(K_CHEB * HIDDEN * HIDDEN + 255) / 256, 256, 0, stream>>>(W2, Wbf2, HIDDEN);

    // ----- normalization: deg -> dis -> norm -----
    zero_f32<<<EW_BLOCKS, 256, 0, stream>>>(dis, (size_t)N_NODES);
    count_deg<<<N_EDGES / 256, 256, 0, stream>>>(row, dis, N_EDGES);
    finalize_dis<<<N_NODES / 256, 256, 0, stream>>>(dis, N_NODES);
    make_norm<<<N_EDGES / 256, 256, 0, stream>>>(row, col, dis, norm, N_EDGES);

    auto launch_gemm = [&](const float* T, const unsigned short* Wk, const float* bl,
                           float* acc, int Cin, int flags) {
        if (Cin == 64) {
            if (flags == GEMM_FLAG_BIAS)
                gemm_cheb<64, GEMM_FLAG_BIAS><<<gemmGrid, 256, 0, stream>>>(T, Wk, bl, acc);
            else if (flags == GEMM_FLAG_RELU)
                gemm_cheb<64, GEMM_FLAG_RELU><<<gemmGrid, 256, 0, stream>>>(T, Wk, bl, acc);
            else
                gemm_cheb<64, 0><<<gemmGrid, 256, 0, stream>>>(T, Wk, bl, acc);
        } else {
            if (flags == GEMM_FLAG_BIAS)
                gemm_cheb<128, GEMM_FLAG_BIAS><<<gemmGrid, 256, 0, stream>>>(T, Wk, bl, acc);
            else if (flags == GEMM_FLAG_RELU)
                gemm_cheb<128, GEMM_FLAG_RELU><<<gemmGrid, 256, 0, stream>>>(T, Wk, bl, acc);
            else
                gemm_cheb<128, 0><<<gemmGrid, 256, 0, stream>>>(T, Wk, bl, acc);
        }
    };

    // one Chebyshev layer: acc = relu?( sum_k T_k @ Wk + b )
    auto run_layer = [&](const float* h, int Cin, const unsigned short* Wbf, const float* bl,
                         float* acc, bool relu) {
        const size_t nT      = (size_t)N_NODES * Cin;
        const size_t Wstride = (size_t)HIDDEN * Cin;   // bf16 elements per k-term

        // k = 0 : T0 = h
        launch_gemm(h, Wbf, bl, acc, Cin, GEMM_FLAG_BIAS);
        // k = 1 : T1 = prop(T0)
        zero_f32<<<EW_BLOCKS, 256, 0, stream>>>(bufA, nT);
        prop_scatter<<<propGrid, 256, 0, stream>>>(row, col, norm, h, bufA, N_EDGES, Cin, 1.0f);
        launch_gemm(bufA, Wbf + Wstride, bl, acc, Cin, 0);
        // k = 2 : T2 = 2*prop(T1) - T0
        neg_copy_f32<<<EW_BLOCKS, 256, 0, stream>>>(h, bufB, nT);
        prop_scatter<<<propGrid, 256, 0, stream>>>(row, col, norm, bufA, bufB, N_EDGES, Cin, 2.0f);
        launch_gemm(bufB, Wbf + 2 * Wstride, bl, acc, Cin, 0);
        // k = 3 : T3 = 2*prop(T2) - T1
        neg_copy_f32<<<EW_BLOCKS, 256, 0, stream>>>(bufA, bufC, nT);
        prop_scatter<<<propGrid, 256, 0, stream>>>(row, col, norm, bufB, bufC, N_EDGES, Cin, 2.0f);
        launch_gemm(bufC, Wbf + 3 * Wstride, bl, acc, Cin, 0);
        // k = 4 : T4 = 2*prop(T3) - T2   (reuse bufA, T1 is dead)
        neg_copy_f32<<<EW_BLOCKS, 256, 0, stream>>>(bufB, bufA, nT);
        prop_scatter<<<propGrid, 256, 0, stream>>>(row, col, norm, bufC, bufA, N_EDGES, Cin, 2.0f);
        launch_gemm(bufA, Wbf + 4 * Wstride, bl, acc, Cin, relu ? GEMM_FLAG_RELU : 0);
    };

    run_layer(x,    F_IN,   Wbf0, b0, bufD, true);   // layer 0: x -> bufD (relu)
    run_layer(bufD, HIDDEN, Wbf1, b1, bufE, true);   // layer 1: bufD -> bufE (relu)
    run_layer(bufE, HIDDEN, Wbf2, b2, bufD, false);  // layer 2: bufE -> bufD (no relu)

    fill_f32<<<(N_GRAPHS * HIDDEN + 255) / 256, 256, 0, stream>>>(gmax, -INFINITY,
                                                                  (size_t)N_GRAPHS * HIDDEN);
    pool_max_partial<<<dim3(N_GRAPHS, NODES_PER_GRAPH / 128), HIDDEN, 0, stream>>>(bufD, gmax);
    head_linear<<<1, 128, 0, stream>>>(gmax, Wlin, blin, out);
}
